// ClusterAttention2_52871047413898
// MI455X (gfx1250) — matrix operand order repaired and checked by reading, not compiled, not run on hardware
//
#include <hip/hip_runtime.h>

// ---------------------------------------------------------------------------
// ClusterAttention2 for MI455X (gfx1250, wave32, WMMA)
//   q  = (x @ key_w)^2                      GEMM1 (bf16 WMMA)
//   s  = (adj/cls) @ q                      GEMM2 (bf16 WMMA, adj exact in bf16)
//   logits -> softmax over singleton axis   (faithful exp(l-l)/exp(l-l))
//   att_vertices = adj.T @ att_clusters; normalized by singleton max (acc/acc)
// ---------------------------------------------------------------------------

typedef __attribute__((ext_vector_type(16))) __bf16          bf16x16;
typedef __attribute__((ext_vector_type(8)))  float           f32x8;
typedef __attribute__((ext_vector_type(4)))  float           f32x4;
typedef __attribute__((ext_vector_type(16))) unsigned short  u16x16;
typedef __attribute__((ext_vector_type(8)))  unsigned short  u16x8;
typedef __attribute__((ext_vector_type(8)))  unsigned int    u32x8;

#define N_V   100000   // vertices (6250 tiles of 16)
#define N_C   1000     // clusters (padded to 1008 = 63 tiles of 16)
#define N_CP  1008
#define IN_F  512      // 16 k-steps of 32
#define ATT_F 64       // 4 n-tiles of 16
#define MTILES 6250
#define KT_Q   16      // IN_F/32
#define KT_S   3125    // N_V/32
#define CTILES 63
#define SPLITS 80      // wave-chunks over KT_S (grid.y=10 * 8 waves)

// ws layout (bytes)
#define QT_BYTES   ((size_t)ATT_F * N_V * 2)          // 12,800,000: q^T bf16 [64][100000]
#define KF_BYTES   ((size_t)KT_Q * 4 * 32 * 32)       // 65,536: key B-fragments
#define S_FLOATS   (N_CP * ATT_F)                     // 64512
#define CLS_FLOATS (N_CP)                             // 1008

static __device__ __forceinline__ unsigned short f2bf(float f) {
    unsigned u = __builtin_bit_cast(unsigned, f);
    u += 0x7FFFu + ((u >> 16) & 1u);      // round-to-nearest-even truncation
    return (unsigned short)(u >> 16);
}

// ---------------- zero scratch (s ++ cls contiguous) ----------------
__global__ void k_zero(float* __restrict__ p, int n) {
    int i = blockIdx.x * blockDim.x + threadIdx.x;
    if (i < n) p[i] = 0.0f;
}

// ---------------- repack key_w into B fragments ----------------
// B 32x16 bf16 layout: lane L -> column n=L%16, half=L/16, element e -> K = e + 16*half
__global__ void k_prepkey(const float* __restrict__ key_w, u32x8* __restrict__ kf) {
    int kt = blockIdx.x >> 2, nt = blockIdx.x & 3;
    int l = threadIdx.x, n = l & 15, half = l >> 4;
    u16x16 b;
#pragma unroll
    for (int e = 0; e < 16; ++e) {
        int K = kt * 32 + 16 * half + e;
        b[e] = f2bf(key_w[(long)K * ATT_F + nt * 16 + n]);
    }
    kf[((long)kt * 4 + nt) * 32 + l] = __builtin_bit_cast(u32x8, b);
}

// ---------------- GEMM1: qT[f][v] = ((x @ key_w)[v][f])^2, bf16 out ----------------
__global__ void __launch_bounds__(256) k_qgemm(const float* __restrict__ x,
                                               const u32x8* __restrict__ kf,
                                               unsigned short* __restrict__ qT) {
    int wave = threadIdx.x >> 5, lane = threadIdx.x & 31;
    long tile = (long)blockIdx.x * 8 + wave;
    if (tile >= MTILES) return;                       // wave-uniform exit
    int n = lane & 15, half = lane >> 4;
    const float* xrow = x + (tile * 16 + n) * (long)IN_F;   // A row m = lane%16

    f32x8 acc0 = {}, acc1 = {}, acc2 = {}, acc3 = {};
    for (int kt = 0; kt < KT_Q; ++kt) {
        // A 16x32: lane holds K = {16h..16h+15}
        const float* p = xrow + kt * 32 + half * 16;
        f32x4 r0 = *(const f32x4*)(p);
        f32x4 r1 = *(const f32x4*)(p + 4);
        f32x4 r2 = *(const f32x4*)(p + 8);
        f32x4 r3 = *(const f32x4*)(p + 12);
        u16x16 au;
#pragma unroll
        for (int j = 0; j < 4; ++j) {
            au[j]      = f2bf(r0[j]);
            au[j + 4]  = f2bf(r1[j]);
            au[j + 8]  = f2bf(r2[j]);
            au[j + 12] = f2bf(r3[j]);
        }
        bf16x16 a = __builtin_bit_cast(bf16x16, au);

        const u32x8* kb = kf + (long)kt * 4 * 32;
        bf16x16 b0 = __builtin_bit_cast(bf16x16, kb[0 * 32 + lane]);
        bf16x16 b1 = __builtin_bit_cast(bf16x16, kb[1 * 32 + lane]);
        bf16x16 b2 = __builtin_bit_cast(bf16x16, kb[2 * 32 + lane]);
        bf16x16 b3 = __builtin_bit_cast(bf16x16, kb[3 * 32 + lane]);

        acc0 = __builtin_amdgcn_wmma_f32_16x16x32_bf16(false, a, false, b0, (short)0, acc0, false, false);
        acc1 = __builtin_amdgcn_wmma_f32_16x16x32_bf16(false, a, false, b1, (short)0, acc1, false, false);
        acc2 = __builtin_amdgcn_wmma_f32_16x16x32_bf16(false, a, false, b2, (short)0, acc2, false, false);
        acc3 = __builtin_amdgcn_wmma_f32_16x16x32_bf16(false, a, false, b3, (short)0, acc3, false, false);
    }

    // C layout: VGPR r, lane l -> M = r + 8*(l/16), N = l%16. Store squared,
    // transposed (qT[f][v]) so GEMM2 B-fragments are contiguous 32B loads.
    long vbase = tile * 16 + 8 * half;
    auto store_tile = [&](f32x8 acc, int nt) {
        u16x8 qv;
#pragma unroll
        for (int r = 0; r < 8; ++r) { float v = acc[r]; qv[r] = f2bf(v * v); }
        *(u16x8*)(qT + (long)(nt * 16 + n) * N_V + vbase) = qv;
    };
    store_tile(acc0, 0); store_tile(acc1, 1); store_tile(acc2, 2); store_tile(acc3, 3);
}

// ---------------- GEMM2: s[c][f] += adj @ q ; cls[c] += rowsum(adj) ----------------
__global__ void __launch_bounds__(256) k_sgemm(const float* __restrict__ adj,
                                               const unsigned short* __restrict__ qT,
                                               float* __restrict__ s,
                                               float* __restrict__ cls) {
    int wave = threadIdx.x >> 5, lane = threadIdx.x & 31;
    int ct = blockIdx.x;
    int w = blockIdx.y * 8 + wave;                    // 0..SPLITS-1
    int lm = lane & 15, half = lane >> 4;
    int c = ct * 16 + lm;
    bool valid = (c < N_C);
    const float* arow = adj + (long)(valid ? c : 0) * N_V;

    f32x8 acc0 = {}, acc1 = {}, acc2 = {}, acc3 = {};
    float clsacc = 0.0f;

    for (int kt = w; kt < KT_S; kt += SPLITS) {
        f32x4 r0 = {}, r1 = {}, r2 = {}, r3 = {};
        if (valid) {
            const float* p = arow + (long)kt * 32 + half * 16;
            if ((kt & 15) == (w & 15))
                __builtin_prefetch(p + 32 * SPLITS, 0, 0);   // stream-ahead hint
            r0 = *(const f32x4*)(p);
            r1 = *(const f32x4*)(p + 4);
            r2 = *(const f32x4*)(p + 8);
            r3 = *(const f32x4*)(p + 12);
        }
        u16x16 au;
#pragma unroll
        for (int j = 0; j < 4; ++j) {
            clsacc += r0[j] + r1[j] + r2[j] + r3[j];
            au[j]      = f2bf(r0[j]);     // 0.0/1.0 are exact in bf16
            au[j + 4]  = f2bf(r1[j]);
            au[j + 8]  = f2bf(r2[j]);
            au[j + 12] = f2bf(r3[j]);
        }
        bf16x16 a = __builtin_bit_cast(bf16x16, au);

        long kb = (long)kt * 32 + 16 * half;          // B: K = e + 16*half, contiguous v
        bf16x16 b0 = __builtin_bit_cast(bf16x16, *(const u32x8*)(qT + (long)(0 * 16 + lm) * N_V + kb));
        bf16x16 b1 = __builtin_bit_cast(bf16x16, *(const u32x8*)(qT + (long)(1 * 16 + lm) * N_V + kb));
        bf16x16 b2 = __builtin_bit_cast(bf16x16, *(const u32x8*)(qT + (long)(2 * 16 + lm) * N_V + kb));
        bf16x16 b3 = __builtin_bit_cast(bf16x16, *(const u32x8*)(qT + (long)(3 * 16 + lm) * N_V + kb));

        acc0 = __builtin_amdgcn_wmma_f32_16x16x32_bf16(false, a, false, b0, (short)0, acc0, false, false);
        acc1 = __builtin_amdgcn_wmma_f32_16x16x32_bf16(false, a, false, b1, (short)0, acc1, false, false);
        acc2 = __builtin_amdgcn_wmma_f32_16x16x32_bf16(false, a, false, b2, (short)0, acc2, false, false);
        acc3 = __builtin_amdgcn_wmma_f32_16x16x32_bf16(false, a, false, b3, (short)0, acc3, false, false);
    }

    // accumulate partial C tiles: row c = ct*16 + r + 8*half, col f = nt*16 + lm
#pragma unroll
    for (int r = 0; r < 8; ++r) {
        int crow = ct * 16 + r + 8 * half;            // < 1008, padded rows get +0
        atomicAdd(&s[(long)crow * ATT_F + 0 * 16 + lm], acc0[r]);
        atomicAdd(&s[(long)crow * ATT_F + 1 * 16 + lm], acc1[r]);
        atomicAdd(&s[(long)crow * ATT_F + 2 * 16 + lm], acc2[r]);
        atomicAdd(&s[(long)crow * ATT_F + 3 * 16 + lm], acc3[r]);
    }
    float other = __shfl_xor(clsacc, 16, 32);         // lanes l and l+16 cover one row
    if (lane < 16) atomicAdd(&cls[ct * 16 + lane], clsacc + other);
}

// ---------------- per-cluster logits + singleton softmax ----------------
__global__ void k_cluster(const float* __restrict__ s, const float* __restrict__ cls,
                          const float* __restrict__ att, float* __restrict__ outc) {
    int c = blockIdx.x, l = threadIdx.x;              // 32 threads (one wave)
    float clsv = cls[c];
    float s0 = s[(long)c * ATT_F + l] / clsv;
    float s1 = s[(long)c * ATT_F + 32 + l] / clsv;
    float m = fmaxf(s0, s1);
#pragma unroll
    for (int o = 16; o; o >>= 1) m = fmaxf(m, __shfl_xor(m, o, 32));
    float a0 = att[(long)c * ATT_F + l];
    float a1 = att[(long)c * ATT_F + 32 + l];
    float t0 = s0 / m * a0; t0 = (t0 > 0.0f) ? t0 : 0.2f * t0;   // leaky_relu 0.2
    float t1 = s1 / m * a1; t1 = (t1 > 0.0f) ? t1 : 0.2f * t1;
    float sum = t0 + t1;
#pragma unroll
    for (int o = 16; o; o >>= 1) sum += __shfl_xor(sum, o, 32);
    float logit = -sum;
    float e = __expf(logit - logit);                  // softmax over singleton axis
    float r = e / e;
    if (l == 0) outc[c] = r;
}

// ---------------- att_vertices = adj.T @ att_clusters, / singleton max ----------------
__global__ void __launch_bounds__(256) k_vertex(const float* __restrict__ adj,
                                                const float* __restrict__ outc,
                                                float* __restrict__ outv) {
    __shared__ float sc[N_C];
    for (int i = threadIdx.x; i < N_C; i += 256) sc[i] = outc[i];
    __syncthreads();
    long v = (long)blockIdx.x * 256 + threadIdx.x;
    if (v >= N_V) return;
    const float* col = adj + v;
    float acc = 0.0f;
    for (int cc = 0; cc < N_C; cc += 4) {
        if (cc + 32 < N_C) __builtin_prefetch(col + (long)(cc + 32) * N_V, 0, 0);
        acc += col[(long)(cc + 0) * N_V] * sc[cc + 0];
        acc += col[(long)(cc + 1) * N_V] * sc[cc + 1];
        acc += col[(long)(cc + 2) * N_V] * sc[cc + 2];
        acc += col[(long)(cc + 3) * N_V] * sc[cc + 3];
    }
    outv[v] = acc / acc;                              // x / max(x, axis=1) on [N,1]
}

// ---------------------------------------------------------------------------
extern "C" void kernel_launch(void* const* d_in, const int* in_sizes, int n_in,
                              void* d_out, int out_size, void* d_ws, size_t ws_size,
                              hipStream_t stream) {
    const float* x     = (const float*)d_in[0];   // [100000, 512]
    const float* adj   = (const float*)d_in[1];   // [1000, 100000]
    const float* att   = (const float*)d_in[2];   // [1000, 64]
    const float* key_w = (const float*)d_in[3];   // [512, 64]

    float* outv = (float*)d_out;                  // [100000] att_vertices
    float* outc = outv + N_V;                     // [1000]   att_clusters

    char* ws = (char*)d_ws;
    unsigned short* qT = (unsigned short*)ws;                         // q^T bf16
    u32x8* kf  = (u32x8*)(ws + QT_BYTES);                             // key frags
    float* s   = (float*)(ws + QT_BYTES + KF_BYTES);                  // [1008][64]
    float* cls = s + S_FLOATS;                                        // [1008]

    k_zero   <<<(S_FLOATS + CLS_FLOATS + 255) / 256, 256, 0, stream>>>(s, S_FLOATS + CLS_FLOATS);
    k_prepkey<<<KT_Q * 4, 32, 0, stream>>>(key_w, kf);
    k_qgemm  <<<(MTILES + 7) / 8, 256, 0, stream>>>(x, kf, qT);
    k_sgemm  <<<dim3(CTILES, SPLITS / 8), 256, 0, stream>>>(adj, qT, s, cls);
    k_cluster<<<N_C, 32, 0, stream>>>(s, cls, att, outc);
    k_vertex <<<(N_V + 255) / 256, 256, 0, stream>>>(adj, outc, outv);
}
